// DcnBlock_20667382628928
// MI455X (gfx1250) — compile-verified, matrix operand order checked by reading, not run on hardware
//
#include <hip/hip_runtime.h>
#include <math.h>

// ---------- types ----------
typedef _Float16 half8 __attribute__((ext_vector_type(8)));
typedef _Float16 v16h  __attribute__((ext_vector_type(16)));
typedef float    v8f   __attribute__((ext_vector_type(8)));
typedef unsigned int uint4v __attribute__((ext_vector_type(4)));
typedef int          int8v  __attribute__((ext_vector_type(8)));
typedef int          int4v  __attribute__((ext_vector_type(4)));

union V16 { v16h v; half8 h8[2]; };

__device__ __forceinline__ V16 load_frag(const _Float16* __restrict__ row, int hi) {
    // halves 0..7  : K = hi*8 + 0..7
    // halves 8..15 : K = 16 + hi*8 + 0..7
    V16 f;
    f.h8[0] = *(const half8*)(row + hi * 8);
    f.h8[1] = *(const half8*)(row + 16 + hi * 8);
    return f;
}

__device__ __forceinline__ v8f wmma16(const V16& a, const V16& b, v8f c) {
    return __builtin_amdgcn_wmma_f32_16x16x32_f16(false, a.v, false, b.v,
                                                  (short)0, c, false, false);
}

// ---------- problem constants ----------
constexpr int B_   = 4;
constexpr int CIN  = 256;
constexpr int CB   = 64;
constexpr int H_   = 112;
constexpr int W_   = 112;
constexpr int HW   = H_ * W_;          // 12544
constexpr int PT   = HW / 16;          // 784 16-pixel tiles per image
constexpr float EPS = 1e-5f;

// ---------- K0: weight conversion / reordering (f32 -> f16, tap-major K) ----------
__global__ void k_prep(const float* __restrict__ w1, const float* __restrict__ woff,
                       const float* __restrict__ w2, const float* __restrict__ w3,
                       _Float16* __restrict__ w1h, _Float16* __restrict__ woffh,
                       _Float16* __restrict__ w2h, _Float16* __restrict__ w3h) {
    int i = blockIdx.x * blockDim.x + threadIdx.x;   // up to 36864
    if (i < CB * CIN)  w1h[i] = (_Float16)w1[i];                       // [64][256]
    if (i < CIN * CB)  w3h[i] = (_Float16)w3[i];                       // [256][64]
    if (i < 32 * 576) {                                                // [32][576], rows 27..31 = 0
        int o = i / 576, k = i % 576, tap = k / 64, c = k % 64;
        woffh[i] = (o < 27) ? (_Float16)woff[(o * 64 + c) * 9 + tap] : (_Float16)0.f;
    }
    if (i < 64 * 576) {                                                // [64][576]
        int co = i / 576, k = i % 576, tap = k / 64, c = k % 64;
        w2h[i] = (_Float16)w2[co * 576 + c * 9 + tap];
    }
}

// ---------- K1: TDM-staged 1x1 conv 256->64 + BN1 + ReLU, WMMA f16 ----------
// One 2D tensor-DMA per block pulls x tile [256 ch][64 pix] (f32, row stride HW)
// straight into LDS; fragments are assembled with on-the-fly f32->f16 conversion.
__global__ __launch_bounds__(128) void k_gemm1(const float* __restrict__ x,
                                               const _Float16* __restrict__ w1h,
                                               const float* __restrict__ g1, const float* __restrict__ b1,
                                               const float* __restrict__ m1, const float* __restrict__ v1,
                                               _Float16* __restrict__ hout) {
    __shared__ __align__(16) float xt[CIN * 64];     // [c][p], 64 KB
    int blk = blockIdx.x;                            // 4 * 196
    int b = blk / 196, p0 = (blk % 196) * 64;
    int tid = threadIdx.x;

#if __has_builtin(__builtin_amdgcn_tensor_load_to_lds)
    if (tid < 32) {
        const float* gptr = x + (long)b * CIN * HW + p0;      // tile start (c=0, p=p0)
        unsigned long long ga = (unsigned long long)(uintptr_t)gptr;
        unsigned ldsb = (unsigned)(uintptr_t)&xt[0];
        // D# group0: count=1 | lds_addr | global_addr[56:0] | type=2
        uint4v g0;
        g0[0] = 1u;
        g0[1] = ldsb;
        g0[2] = (unsigned)(ga & 0xFFFFFFFFu);
        g0[3] = (unsigned)((ga >> 32) & 0x01FFFFFFull) | (2u << 30);
        // D# group1: wg_mask=0, data_size=2 (4B), tensor_dim0=64, tensor_dim1=256,
        //            tile_dim0=64, tile_dim1=256, tile_dim2=0, dim0_stride=HW, dim1_stride=0
        int8v g1v;
        g1v[0] = (2 << 16);          // data_size = 4 bytes
        g1v[1] = (64 << 16);         // tensor_dim0[15:0] in bits 63:48
        g1v[2] = (256 << 16);        // tensor_dim0[31:16]=0 | tensor_dim1[15:0]
        g1v[3] = (64 << 16);         // tensor_dim1[31:16]=0 | tile_dim0=64
        g1v[4] = 256;                // tile_dim1=256 | tile_dim2=0
        g1v[5] = HW;                 // tensor_dim0_stride[31:0]
        g1v[6] = 0;                  // stride[47:32]=0 | dim1_stride[15:0]=0
        g1v[7] = 0;
        int4v gz4 = {0, 0, 0, 0};                    // 2D tile: groups 2/3 unused
        int8v gz8 = {0, 0, 0, 0, 0, 0, 0, 0};       // extra group (6-arg toolchain): null
        __builtin_amdgcn_tensor_load_to_lds(g0, g1v, gz4, gz4, gz8, 0);
        __builtin_amdgcn_s_wait_tensorcnt(0);
    }
    __syncthreads();
#else
    for (int i = tid; i < CIN * 64; i += 128)
        xt[i] = x[((long)b * CIN + (i >> 6)) * HW + p0 + (i & 63)];
    __syncthreads();
#endif

    int lane = tid & 31, wid = tid >> 5;
    int r = lane & 15, hi = lane >> 4;
    int pl = wid * 16 + r;                           // this lane's pixel within the 64-tile

    v8f acc[4] = {};
#pragma unroll
    for (int kc = 0; kc < 8; ++kc) {
        int kb = kc * 32;
        V16 a;                                       // conflict-free: lanes walk consecutive pixels
#pragma unroll
        for (int j = 0; j < 8; ++j) {
            a.h8[0][j] = (_Float16)xt[(kb + hi * 8 + j) * 64 + pl];
            a.h8[1][j] = (_Float16)xt[(kb + 16 + hi * 8 + j) * 64 + pl];
        }
#pragma unroll
        for (int n = 0; n < 4; ++n) {
            V16 bb = load_frag(w1h + (n * 16 + r) * CIN + kb, hi);
            acc[n] = wmma16(a, bb, acc[n]);
        }
    }
    long pbase = (long)b * HW + p0 + wid * 16;
#pragma unroll
    for (int n = 0; n < 4; ++n) {
        int ch = n * 16 + r;
        float s = g1[ch] * rsqrtf(v1[ch] + EPS);
        float t = b1[ch] - m1[ch] * s;
#pragma unroll
        for (int v = 0; v < 8; ++v) {
            float val = fmaxf(acc[n][v] * s + t, 0.f);
            hout[(pbase + v + 8 * hi) * CB + ch] = (_Float16)val;
        }
    }
}

// ---------- K2: fused offset-conv + bilinear sampling + deform-conv + BN2 + ReLU ----------
__global__ __launch_bounds__(32) void k_dcn(const _Float16* __restrict__ hf,
                                            const _Float16* __restrict__ woffh,
                                            const _Float16* __restrict__ w2h,
                                            const float* __restrict__ boff, const float* __restrict__ dcnb,
                                            const float* __restrict__ g2, const float* __restrict__ b2,
                                            const float* __restrict__ m2, const float* __restrict__ v2,
                                            _Float16* __restrict__ gout) {
    __shared__ float offbuf[16][32];
    __shared__ __align__(16) _Float16 S[16][592];   // 576 + 16 pad, rows 16B-aligned

    int tile = blockIdx.x;                    // 4 * 112 * 7
    int lane = threadIdx.x, r = lane & 15, hi = lane >> 4;
    int b = tile / PT;
    int rem = tile % PT;
    int y = rem / 7, x0 = (rem % 7) * 16;

    // ---- stage A: offset conv  off[16 pix][27] = h_patch[16][576] * woffh^T ----
    v8f oc[2] = {};
#pragma unroll
    for (int kc = 0; kc < 18; ++kc) {
        int tap = kc >> 1;
        int dy = tap / 3 - 1, dx = tap % 3 - 1;
        int px = x0 + r, yy = y + dy, xx = px + dx;
        bool valid = (yy >= 0) & (yy < H_) & (xx >= 0) & (xx < W_);
        int yyc = min(max(yy, 0), H_ - 1), xxc = min(max(xx, 0), W_ - 1);
        int cb = (kc & 1) * 32;               // tap*64 + cb == kc*32
        V16 a = load_frag(hf + (((long)b * HW + yyc * W_ + xxc) * CB + cb), hi);
        if (!valid) { V16 z = {}; a = z; }    // zero-padding via lane VALU, EXEC stays full
#pragma unroll
        for (int n = 0; n < 2; ++n) {
            V16 bb = load_frag(woffh + (n * 16 + r) * 576 + kc * 32, hi);
            oc[n] = wmma16(a, bb, oc[n]);
        }
    }
#pragma unroll
    for (int n = 0; n < 2; ++n) {
        int o = n * 16 + r;
        float bo = (o < 27) ? boff[o] : 0.f;
#pragma unroll
        for (int v = 0; v < 8; ++v) offbuf[v + 8 * hi][o] = oc[n][v] + bo;
    }
    __syncthreads();

    // ---- stage B: modulated bilinear sampling -> S[pix][tap*64+c] (f16 in LDS) ----
    for (int it = 0; it < 5; ++it) {
        int idx = it * 32 + lane;             // 144 (pix,tap) pairs
        if (idx < 144) {
            int pix = idx / 9, tap = idx % 9;
            float offy = offbuf[pix][2 * tap];
            float offx = offbuf[pix][2 * tap + 1];
            float ml   = offbuf[pix][18 + tap];
            float msk  = 1.f / (1.f + __expf(-ml));
            int px = x0 + pix;
            float sy = offy + (float)(y  - 1 + tap / 3);
            float sx = offx + (float)(px - 1 + tap % 3);
            float fy = floorf(sy), fx = floorf(sx);
            int iy0 = (int)fy, ix0 = (int)fx;
            float wy1 = sy - fy, wx1 = sx - fx;
            float wy0 = 1.f - wy1, wx0 = 1.f - wx1;
            float w00 = wy0 * wx0, w01 = wy0 * wx1, w10 = wy1 * wx0, w11 = wy1 * wx1;
            bool vy0 = (iy0 >= 0) & (iy0 < H_), vy1 = (iy0 + 1 >= 0) & (iy0 + 1 < H_);
            bool vx0 = (ix0 >= 0) & (ix0 < W_), vx1 = (ix0 + 1 >= 0) & (ix0 + 1 < W_);
            w00 *= (vy0 & vx0); w01 *= (vy0 & vx1); w10 *= (vy1 & vx0); w11 *= (vy1 & vx1);
            int y0c = min(max(iy0, 0), H_ - 1), y1c = min(max(iy0 + 1, 0), H_ - 1);
            int x0c = min(max(ix0, 0), W_ - 1), x1c = min(max(ix0 + 1, 0), W_ - 1);
            const _Float16* p00 = hf + ((long)b * HW + y0c * W_ + x0c) * CB;
            const _Float16* p01 = hf + ((long)b * HW + y0c * W_ + x1c) * CB;
            const _Float16* p10 = hf + ((long)b * HW + y1c * W_ + x0c) * CB;
            const _Float16* p11 = hf + ((long)b * HW + y1c * W_ + x1c) * CB;
#pragma unroll
            for (int c = 0; c < CB; c += 8) {
                half8 a0 = *(const half8*)(p00 + c);
                half8 a1 = *(const half8*)(p01 + c);
                half8 a2 = *(const half8*)(p10 + c);
                half8 a3 = *(const half8*)(p11 + c);
                half8 rr;
#pragma unroll
                for (int j = 0; j < 8; ++j) {
                    float s = w00 * (float)a0[j] + w01 * (float)a1[j]
                            + w10 * (float)a2[j] + w11 * (float)a3[j];
                    rr[j] = (_Float16)(msk * s);
                }
                *(half8*)(&S[pix][tap * CB + c]) = rr;
            }
        }
    }
    __syncthreads();

    // ---- stage C: deform conv  out[16][64] = S[16][576] * w2h^T ----
    v8f mc[4] = {};
#pragma unroll
    for (int kc = 0; kc < 18; ++kc) {
        int kb = kc * 32;
        V16 a = load_frag(&S[r][kb], hi);
#pragma unroll
        for (int n = 0; n < 4; ++n) {
            V16 bb = load_frag(w2h + (n * 16 + r) * 576 + kb, hi);
            mc[n] = wmma16(a, bb, mc[n]);
        }
    }
    long p0 = (long)b * HW + y * W_ + x0;
#pragma unroll
    for (int n = 0; n < 4; ++n) {
        int ch = n * 16 + r;
        float s = g2[ch] * rsqrtf(v2[ch] + EPS);
        float t = b2[ch] - m2[ch] * s + dcnb[ch] * s;   // BN2(conv + dcn_b)
#pragma unroll
        for (int v = 0; v < 8; ++v) {
            float val = fmaxf(mc[n][v] * s + t, 0.f);
            gout[(p0 + v + 8 * hi) * CB + ch] = (_Float16)val;
        }
    }
}

// ---------- K3: 1x1 conv 64->256 + BN3 + residual + ReLU ----------
__global__ __launch_bounds__(32) void k_gemm3(const _Float16* __restrict__ g,
                                              const _Float16* __restrict__ w3h,
                                              const float* __restrict__ x,
                                              const float* __restrict__ g3, const float* __restrict__ b3,
                                              const float* __restrict__ m3, const float* __restrict__ v3,
                                              float* __restrict__ out) {
    int blk = blockIdx.x;                    // 3136 pixel-tiles * 4 cout-quarters
    int nq = blk & 3, tile = blk >> 2;
    int lane = threadIdx.x, r = lane & 15, hi = lane >> 4;
    int b = tile / PT;
    int pl0 = (tile % PT) * 16;              // pixel within image
    long p0 = (long)b * HW + pl0;
    const _Float16* arow = g + (p0 + r) * CB;

    v8f acc[4] = {};
#pragma unroll
    for (int kc = 0; kc < 2; ++kc) {
        int kb = kc * 32;
        V16 a = load_frag(arow + kb, hi);
#pragma unroll
        for (int n = 0; n < 4; ++n) {
            int ch = nq * 64 + n * 16 + r;
            V16 bb = load_frag(w3h + ch * CB + kb, hi);
            acc[n] = wmma16(a, bb, acc[n]);
        }
    }
#pragma unroll
    for (int n = 0; n < 4; ++n) {
        int ch = nq * 64 + n * 16 + r;
        float s = g3[ch] * rsqrtf(v3[ch] + EPS);
        float t = b3[ch] - m3[ch] * s;
#pragma unroll
        for (int v = 0; v < 8; ++v) {
            int pl = pl0 + v + 8 * hi;
            long gi = ((long)b * CIN + ch) * HW + pl;
            float val = fmaxf(acc[n][v] * s + t + x[gi], 0.f);
            out[gi] = val;
        }
    }
}

// ---------- launcher ----------
extern "C" void kernel_launch(void* const* d_in, const int* in_sizes, int n_in,
                              void* d_out, int out_size, void* d_ws, size_t ws_size,
                              hipStream_t stream) {
    const float* x    = (const float*)d_in[0];
    const float* w1   = (const float*)d_in[1];
    const float* bn1g = (const float*)d_in[2];
    const float* bn1b = (const float*)d_in[3];
    const float* bn1m = (const float*)d_in[4];
    const float* bn1v = (const float*)d_in[5];
    const float* woff = (const float*)d_in[6];
    const float* boff = (const float*)d_in[7];
    const float* w2   = (const float*)d_in[8];
    const float* dcnb = (const float*)d_in[9];
    const float* bn2g = (const float*)d_in[10];
    const float* bn2b = (const float*)d_in[11];
    const float* bn2m = (const float*)d_in[12];
    const float* bn2v = (const float*)d_in[13];
    const float* w3   = (const float*)d_in[14];
    const float* bn3g = (const float*)d_in[15];
    const float* bn3b = (const float*)d_in[16];
    const float* bn3m = (const float*)d_in[17];
    const float* bn3v = (const float*)d_in[18];
    float* out = (float*)d_out;

    char* ws = (char*)d_ws;
    size_t off = 0;
    _Float16* hf    = (_Float16*)(ws + off); off += (size_t)B_ * HW * CB * 2;   // 6.4 MB
    _Float16* gbuf  = (_Float16*)(ws + off); off += (size_t)B_ * HW * CB * 2;   // 6.4 MB
    _Float16* w1h   = (_Float16*)(ws + off); off += (size_t)CB * CIN * 2;
    _Float16* woffh = (_Float16*)(ws + off); off += (size_t)32 * 576 * 2;
    _Float16* w2h   = (_Float16*)(ws + off); off += (size_t)CB * 576 * 2;
    _Float16* w3h   = (_Float16*)(ws + off); off += (size_t)CIN * CB * 2;

    k_prep<<<144, 256, 0, stream>>>(w1, woff, w2, w3, w1h, woffh, w2h, w3h);
    k_gemm1<<<784, 128, 0, stream>>>(x, w1h, bn1g, bn1b, bn1m, bn1v, hf);
    k_dcn<<<3136, 32, 0, stream>>>(hf, woffh, w2h, boff, dcnb,
                                   bn2g, bn2b, bn2m, bn2v, gbuf);
    k_gemm3<<<12544, 32, 0, stream>>>(gbuf, w3h, x, bn3g, bn3b, bn3m, bn3v, out);
}
